// GINEdge_90898687852685
// MI455X (gfx1250) — matrix operand order repaired and checked by reading, not compiled, not run on hardware
//
#include <hip/hip_runtime.h>
#include <hip/hip_bf16.h>

// ---------------------------------------------------------------------------
// GIN (edge-featured) forward for MI455X / gfx1250.
// Big GEMMs -> v_wmma_f32_16x16x32_bf16. Full B-panel (<=128KB) staged once
// into LDS via global_load_async_to_lds_b128 (ASYNCcnt), single barrier,
// barrier-free K-loop with software-pipelined A streaming.
// Graph scatter/gather + BN + pooling -> fp32 atomics / elementwise kernels.
// ---------------------------------------------------------------------------

#define NNODES 50000
#define NEDGES 800000
#define NGRAPH 1024
#define HID    256
#define FNODE  141
#define FEDGE  12
#define NTASK  12
#define NLAYER 5

typedef __bf16 v16bf __attribute__((ext_vector_type(16)));
typedef __bf16 v8bf  __attribute__((ext_vector_type(8)));
typedef float  v8f   __attribute__((ext_vector_type(8)));

union BF16x16 { v16bf v; v8bf h[2]; };

static inline int cdiv(int a, int b) { return (a + b - 1) / b; }

// --------------------------- WMMA fragment loads ---------------------------

// A row-major (M x Kp). Row index clamped (OOB rows dropped at the store).
__device__ inline v16bf load_a_frag(const __bf16* __restrict__ A, int M, int lda,
                                    int m0, int k0, int lane) {
  const int l  = lane & 15;
  const int hi = lane >> 4;           // lanes 16-31 hold K+16 .. K+31
  const int m  = min(m0 + l, M - 1);
  BF16x16 a;
  const v8bf* p = (const v8bf*)(A + (size_t)m * lda + k0 + hi * 16);
  a.h[0] = p[0];                      // K = k0+hi*16   .. +7
  a.h[1] = p[1];                      // K = k0+hi*16+8 .. +15
  return a.v;
}

// B panel resident in LDS, N-major with row stride Kp (bf16 elements).
__device__ inline v16bf load_b_frag_lds(const __bf16* Bs, int ldb, int n_loc,
                                        int k0, int lane) {
  const int n  = n_loc + (lane & 15);
  const int ks = k0 + (lane >> 4) * 16;
  return *(const v16bf*)(Bs + (size_t)n * ldb + ks);   // 32B -> ds_load_b128 x2
}

// ------------------------- async B-panel staging ---------------------------

__device__ inline void async_b128(const __bf16* gsrc, unsigned lds_off) {
  asm volatile("global_load_async_to_lds_b128 %0, %1, off"
               :: "v"(lds_off), "v"(gsrc) : "memory");
}
__device__ inline void wait_async0() {
  asm volatile("s_wait_asynccnt 0x0" ::: "memory");
}

// ------------------------------- GEMM kernel -------------------------------
// C[M,Np] = A[M,Kp](bf16) * Wt[Np,Kp]^T(bf16) + bias
// Block: 256 thr = 8 waves as 4(m) x 2(n); block tile 128(M) x 128(N);
// wave tile 32 x 64. Whole 128 x Kp B panel staged in LDS up-front.
// mode 0: store fp32    mode 1: store bf16
// mode 2: atomic scatter-add rows into scatDst[scatIdx[m]] (fp32)
__global__ __launch_bounds__(256)
void gemm_bf16_wmma(const __bf16* __restrict__ A, const __bf16* __restrict__ Wt,
                    const float* __restrict__ bias,
                    float* __restrict__ outF, __bf16* __restrict__ outBF,
                    float* __restrict__ scatDst, const int* __restrict__ scatIdx,
                    int M, int Kp, int Np, int mode) {
  extern __shared__ __align__(32) char smem[];
  __bf16* Bs = (__bf16*)smem;                 // 128 * Kp bf16

  const int t    = threadIdx.x;
  const int lane = t & 31;
  const int wave = t >> 5;
  const int mw   = wave & 3;                  // 4 m-waves
  const int nw   = wave >> 2;                 // 2 n-waves
  const int m0   = blockIdx.x * 128 + mw * 32;
  const int n0   = blockIdx.y * 128;          // panel origin (global col)
  const int nl0  = nw * 64;                   // wave col offset inside panel

  // ---- stage full 128 x Kp B panel (contiguous in Wt) into LDS ----
  {
    const __bf16* panel = Wt + (size_t)n0 * Kp;
    const int sweeps = (128 * Kp * 2) >> 12;  // 4KB per 256-thread sweep
    for (int i = 0; i < sweeps; ++i) {
      const int byte_off = (i << 12) + t * 16;
      async_b128(panel + (byte_off >> 1), (unsigned)(size_t)(smem + byte_off));
    }
    wait_async0();
    __syncthreads();                          // panel resident for all waves
  }

  v8f c[2][4];
  #pragma unroll
  for (int mt = 0; mt < 2; ++mt)
    #pragma unroll
    for (int j = 0; j < 4; ++j) c[mt][j] = (v8f){};

  const int nsteps = Kp >> 5;
  v16bf a0 = load_a_frag(A, M, Kp, m0,      0, lane);
  v16bf a1 = load_a_frag(A, M, Kp, m0 + 16, 0, lane);

  for (int s = 0; s < nsteps; ++s) {
    const int k0 = s << 5;
    // pull all 4 B frags before the WMMA chain (one dscnt wait covers them)
    v16bf b0 = load_b_frag_lds(Bs, Kp, nl0 +  0, k0, lane);
    v16bf b1 = load_b_frag_lds(Bs, Kp, nl0 + 16, k0, lane);
    v16bf b2 = load_b_frag_lds(Bs, Kp, nl0 + 32, k0, lane);
    v16bf b3 = load_b_frag_lds(Bs, Kp, nl0 + 48, k0, lane);
    // prefetch next K-step's A while WMMAs run
    v16bf na0 = a0, na1 = a1;
    if (s + 1 < nsteps) {
      na0 = load_a_frag(A, M, Kp, m0,      k0 + 32, lane);
      na1 = load_a_frag(A, M, Kp, m0 + 16, k0 + 32, lane);
    }
    c[0][0] = __builtin_amdgcn_wmma_f32_16x16x32_bf16(false, a0, false, b0, (short)0, c[0][0], false, false);
    c[1][0] = __builtin_amdgcn_wmma_f32_16x16x32_bf16(false, a1, false, b0, (short)0, c[1][0], false, false);
    c[0][1] = __builtin_amdgcn_wmma_f32_16x16x32_bf16(false, a0, false, b1, (short)0, c[0][1], false, false);
    c[1][1] = __builtin_amdgcn_wmma_f32_16x16x32_bf16(false, a1, false, b1, (short)0, c[1][1], false, false);
    c[0][2] = __builtin_amdgcn_wmma_f32_16x16x32_bf16(false, a0, false, b2, (short)0, c[0][2], false, false);
    c[1][2] = __builtin_amdgcn_wmma_f32_16x16x32_bf16(false, a1, false, b2, (short)0, c[1][2], false, false);
    c[0][3] = __builtin_amdgcn_wmma_f32_16x16x32_bf16(false, a0, false, b3, (short)0, c[0][3], false, false);
    c[1][3] = __builtin_amdgcn_wmma_f32_16x16x32_bf16(false, a1, false, b3, (short)0, c[1][3], false, false);
    a0 = na0; a1 = na1;
  }

  const int l  = lane & 15;
  const int hi = lane >> 4;
  #pragma unroll
  for (int mt = 0; mt < 2; ++mt) {
    #pragma unroll
    for (int j = 0; j < 4; ++j) {
      const int n = n0 + nl0 + 16 * j + l;
      const float bn = bias ? bias[n] : 0.0f;
      #pragma unroll
      for (int v = 0; v < 8; ++v) {
        const int m = m0 + 16 * mt + v + 8 * hi;
        if (m < M) {
          const float val = c[mt][j][v] + bn;
          if (mode == 0) {
            outF[(size_t)m * Np + n] = val;
          } else if (mode == 1) {
            outBF[(size_t)m * Np + n] = (__bf16)val;
          } else {
            const int r = scatIdx[m];
            unsafeAtomicAdd(&scatDst[(size_t)r * Np + n], val);
          }
        }
      }
    }
  }
}

// --------------------------- data-prep / pointwise -------------------------

__global__ void cvt_pad_bf16_k(const float* __restrict__ in, __bf16* __restrict__ out,
                               int M, int K, int Kp) {
  size_t idx = (size_t)blockIdx.x * blockDim.x + threadIdx.x;
  size_t tot = (size_t)M * Kp;
  if (idx >= tot) return;
  int k = (int)(idx % Kp);
  size_t m = idx / Kp;
  out[idx] = (k < K) ? (__bf16)in[m * K + k] : (__bf16)0.0f;
}

__global__ void transpose_w_bf16_k(const float* __restrict__ W, __bf16* __restrict__ Wt,
                                   int K, int N, int Kp, int Np) {
  size_t idx = (size_t)blockIdx.x * blockDim.x + threadIdx.x;
  size_t tot = (size_t)Np * Kp;
  if (idx >= tot) return;
  int k = (int)(idx % Kp);
  int n = (int)(idx / Kp);
  Wt[idx] = (n < N && k < K) ? (__bf16)W[(size_t)k * N + n] : (__bf16)0.0f;
}

__global__ void fill_f32_k(float* __restrict__ p, float v, size_t n) {
  size_t i = (size_t)blockIdx.x * blockDim.x + threadIdx.x;
  if (i < n) p[i] = v;
}

__global__ void add_ew_k(const float* __restrict__ a, const float* __restrict__ b,
                         float* __restrict__ o, size_t n) {
  size_t i = (size_t)blockIdx.x * blockDim.x + threadIdx.x;
  if (i < n) o[i] = a[i] + b[i];
}

// agg[col[e]] += hin[row[e]]  (4 floats per thread)
__global__ void scatter_agg_k(const float* __restrict__ hin, const int* __restrict__ row,
                              const int* __restrict__ col, float* __restrict__ agg, int E) {
  size_t idx = (size_t)blockIdx.x * blockDim.x + threadIdx.x;
  size_t tot = (size_t)E * (HID / 4);
  if (idx >= tot) return;
  int e = (int)(idx / (HID / 4));
  int c = (int)(idx % (HID / 4)) * 4;
  const float4 v = *(const float4*)(hin + (size_t)row[e] * HID + c);
  float* d = agg + (size_t)col[e] * HID + c;
  unsafeAtomicAdd(d + 0, v.x);
  unsafeAtomicAdd(d + 1, v.y);
  unsafeAtomicAdd(d + 2, v.z);
  unsafeAtomicAdd(d + 3, v.w);
}

// g = (1+eps)*hin + agg, emitted as bf16 (feeds next GEMM)
__global__ void combine_gin_k(const float* __restrict__ hin, const float* __restrict__ agg,
                              const float* __restrict__ epsp, __bf16* __restrict__ out, size_t n) {
  size_t i = (size_t)blockIdx.x * blockDim.x + threadIdx.x;
  if (i >= n) return;
  float s = 1.0f + epsp[0];
  out[i] = (__bf16)(s * hin[i] + agg[i]);
}

// column sums + sums of squares (blockDim.x == C)
__global__ void bn_stats_k(const float* __restrict__ X, float* __restrict__ sums,
                           int Mrows, int C, int rowsPerBlock) {
  int c = threadIdx.x;
  int r0 = blockIdx.x * rowsPerBlock;
  int r1 = min(r0 + rowsPerBlock, Mrows);
  float s = 0.0f, q = 0.0f;
  for (int r = r0; r < r1; ++r) {
    float v = X[(size_t)r * C + c];
    s += v; q += v * v;
  }
  unsafeAtomicAdd(&sums[c], s);
  unsafeAtomicAdd(&sums[C + c], q);
}

__global__ void bn_finalize_k(const float* __restrict__ sums, const float* __restrict__ gamma,
                              const float* __restrict__ beta, float* __restrict__ ss,
                              int Mrows, int C) {
  int c = threadIdx.x + blockIdx.x * blockDim.x;
  if (c >= C) return;
  float inv = 1.0f / (float)Mrows;
  float mean = sums[c] * inv;
  float var  = sums[C + c] * inv - mean * mean;
  float sc   = gamma[c] * rsqrtf(var + 1e-5f);
  ss[c]     = sc;
  ss[C + c] = beta[c] - mean * sc;
}

__global__ void bn_apply_k(const float* __restrict__ X, const float* __restrict__ ss,
                           float* __restrict__ outF, __bf16* __restrict__ outBF,
                           int C, size_t n, int relu) {
  size_t i = (size_t)blockIdx.x * blockDim.x + threadIdx.x;
  if (i >= n) return;
  int c = (int)(i % C);
  float v = X[i] * ss[c] + ss[C + c];
  if (relu) v = fmaxf(v, 0.0f);
  if (outF)  outF[i]  = v;
  if (outBF) outBF[i] = (__bf16)v;
}

// ------------------------------- pooling -----------------------------------

__global__ void count_nodes_k(const int* __restrict__ batch, float* __restrict__ counts, int N) {
  int i = blockIdx.x * blockDim.x + threadIdx.x;
  if (i < N) unsafeAtomicAdd(&counts[batch[i]], 1.0f);
}

__global__ void pool_sum_max_k(const float* __restrict__ h, const int* __restrict__ batch,
                               float* __restrict__ psum, unsigned* __restrict__ pmax, int N) {
  size_t idx = (size_t)blockIdx.x * blockDim.x + threadIdx.x;
  size_t tot = (size_t)N * HID;
  if (idx >= tot) return;
  int nn = (int)(idx / HID);
  int c  = (int)(idx % HID);
  int b  = batch[nn];
  float v = h[idx];                      // post-ReLU, >= 0: uint bit order works
  unsafeAtomicAdd(&psum[(size_t)b * HID + c], v);
  atomicMax(&pmax[(size_t)b * HID + c], __float_as_uint(v));
}

__global__ void pool_to_z_k(const float* __restrict__ psum, const unsigned* __restrict__ pmax,
                            const float* __restrict__ counts, __bf16* __restrict__ zbf, int B) {
  size_t idx = (size_t)blockIdx.x * blockDim.x + threadIdx.x;
  size_t tot = (size_t)B * HID;
  if (idx >= tot) return;
  int b = (int)(idx / HID);
  int c = (int)(idx % HID);
  float d = fmaxf(counts[b], 1.0f);
  zbf[(size_t)b * (2 * HID) + c]       = (__bf16)(psum[idx] / d);
  zbf[(size_t)b * (2 * HID) + HID + c] = (__bf16)__uint_as_float(pmax[idx]);
}

__global__ void pad_bias_k(const float* __restrict__ b, float* __restrict__ out, int N, int Np) {
  int i = blockIdx.x * blockDim.x + threadIdx.x;
  if (i < Np) out[i] = (i < N) ? b[i] : 0.0f;
}

__global__ void copy_out_k(const float* __restrict__ inp, float* __restrict__ out,
                           int B, int T, int ldi) {
  int i = blockIdx.x * blockDim.x + threadIdx.x;
  if (i >= B * T) return;
  int b = i / T, t = i % T;
  out[i] = inp[(size_t)b * ldi + t];
}

// ------------------------------- host side ---------------------------------

static void* wsp(char* base, size_t& off, size_t bytes) {
  void* p = base + off;
  off += (bytes + 255) & ~(size_t)255;
  return p;
}

static void launch_gemm(const __bf16* A, const __bf16* Wt, const float* bias,
                        float* outF, __bf16* outBF, float* scatDst, const int* scatIdx,
                        int M, int Kp, int Np, int mode, hipStream_t s) {
  dim3 grid(cdiv(M, 128), Np / 128);
  size_t lds_bytes = (size_t)128 * Kp * 2;       // <= 128KB (Kp <= 512)
  gemm_bf16_wmma<<<grid, 256, lds_bytes, s>>>(A, Wt, bias, outF, outBF,
                                              scatDst, scatIdx, M, Kp, Np, mode);
}

#define EW(n) (int)(((size_t)(n) + 255) / 256), 256, 0, stream

extern "C" void kernel_launch(void* const* d_in, const int* in_sizes, int n_in,
                              void* d_out, int out_size, void* d_ws, size_t ws_size,
                              hipStream_t stream) {
  // ---- inputs (setup_inputs dict order, params flattened in insertion order)
  const float* x      = (const float*)d_in[0];
  const int*   ei     = (const int*)  d_in[1];
  const float* eattr  = (const float*)d_in[2];
  const int*   batch  = (const int*)  d_in[3];
  const float* node_W = (const float*)d_in[4];
  const float* node_b = (const float*)d_in[5];
  const float* edge_W0= (const float*)d_in[6];
  const float* edge_b0= (const float*)d_in[7];
  const float* out_W1 = (const float*)d_in[8 + 11 * NLAYER + 0];
  const float* out_b1 = (const float*)d_in[8 + 11 * NLAYER + 1];
  const float* out_g  = (const float*)d_in[8 + 11 * NLAYER + 2];
  const float* out_be = (const float*)d_in[8 + 11 * NLAYER + 3];
  const float* out_W2 = (const float*)d_in[8 + 11 * NLAYER + 4];
  const float* out_b2 = (const float*)d_in[8 + 11 * NLAYER + 5];
  const int* row = ei;            // edge_index[0]
  const int* col = ei + NEDGES;   // edge_index[1]
  float* out = (float*)d_out;

  // ---- workspace carve-out
  char* base = (char*)d_ws;
  size_t off = 0;
  const int FN_P = 160, FE_P = 32;   // K padded to multiples of 32
  __bf16*   x_bf    = (__bf16*)wsp(base, off, (size_t)NNODES * FN_P * 2);
  __bf16*   ea_in_bf= (__bf16*)wsp(base, off, (size_t)NEDGES * FE_P * 2);
  __bf16*   ea_bf   = (__bf16*)wsp(base, off, (size_t)NEDGES * HID * 2);
  float*    h       = (float*) wsp(base, off, (size_t)NNODES * HID * 4);
  float*    x_edge  = (float*) wsp(base, off, (size_t)NNODES * HID * 4);
  float*    hin     = (float*) wsp(base, off, (size_t)NNODES * HID * 4);
  float*    agg     = (float*) wsp(base, off, (size_t)NNODES * HID * 4);
  float*    tmp1    = (float*) wsp(base, off, (size_t)NNODES * HID * 4);
  __bf16*   g_bf    = (__bf16*)wsp(base, off, (size_t)NNODES * HID * 2);
  __bf16*   Wt      = (__bf16*)wsp(base, off, (size_t)512 * 512 * 2);
  float*    bnsums  = (float*) wsp(base, off, 2 * HID * 4);
  float*    bnss    = (float*) wsp(base, off, 2 * HID * 4);
  float*    counts  = (float*) wsp(base, off, (size_t)NGRAPH * 4);
  float*    psum    = (float*) wsp(base, off, (size_t)NGRAPH * HID * 4);
  unsigned* pmax    = (unsigned*)wsp(base, off, (size_t)NGRAPH * HID * 4);
  __bf16*   z_bf    = (__bf16*)wsp(base, off, (size_t)NGRAPH * 2 * HID * 2);
  float*    y1      = (float*) wsp(base, off, (size_t)NGRAPH * HID * 4);
  __bf16*   zb2     = (__bf16*)wsp(base, off, (size_t)NGRAPH * HID * 2);
  float*    bias128 = (float*) wsp(base, off, 128 * 4);
  float*    outpad  = (float*) wsp(base, off, (size_t)NGRAPH * 128 * 4);
  (void)ws_size; (void)n_in; (void)in_sizes; (void)out_size;

  // ---- encoders -----------------------------------------------------------
  // h = x @ node_W + node_b
  cvt_pad_bf16_k<<<EW((size_t)NNODES * FN_P)>>>(x, x_bf, NNODES, FNODE, FN_P);
  transpose_w_bf16_k<<<EW((size_t)HID * FN_P)>>>(node_W, Wt, FNODE, HID, FN_P, HID);
  launch_gemm(x_bf, Wt, node_b, h, nullptr, nullptr, nullptr, NNODES, FN_P, HID, 0, stream);

  // ea = edge_attr @ edge_W + edge_b  (stored bf16: consumed only by layer GEMMs)
  cvt_pad_bf16_k<<<EW((size_t)NEDGES * FE_P)>>>(eattr, ea_in_bf, NEDGES, FEDGE, FE_P);
  transpose_w_bf16_k<<<EW((size_t)HID * FE_P)>>>(edge_W0, Wt, FEDGE, HID, FE_P, HID);
  launch_gemm(ea_in_bf, Wt, edge_b0, nullptr, ea_bf, nullptr, nullptr, NEDGES, FE_P, HID, 1, stream);

  // ---- GIN layers ---------------------------------------------------------
  const size_t NH = (size_t)NNODES * HID;
  for (int l = 0; l < NLAYER; ++l) {
    const int pb = 8 + 11 * l;
    const float* lW   = (const float*)d_in[pb + 0];
    const float* lb   = (const float*)d_in[pb + 1];
    const float* eps  = (const float*)d_in[pb + 2];
    const float* W1   = (const float*)d_in[pb + 3];
    const float* b1   = (const float*)d_in[pb + 4];
    const float* g1   = (const float*)d_in[pb + 5];
    const float* be1  = (const float*)d_in[pb + 6];
    const float* W2   = (const float*)d_in[pb + 7];
    const float* b2   = (const float*)d_in[pb + 8];
    const float* bng  = (const float*)d_in[pb + 9];
    const float* bnb  = (const float*)d_in[pb + 10];

    // x_edge = scatter_add(ea @ lW + lb, row)   (fused GEMM epilogue atomics)
    fill_f32_k<<<EW(NH)>>>(x_edge, 0.0f, NH);
    transpose_w_bf16_k<<<EW((size_t)HID * HID)>>>(lW, Wt, HID, HID, HID, HID);
    launch_gemm(ea_bf, Wt, lb, nullptr, nullptr, x_edge, row, NEDGES, HID, HID, 2, stream);

    // hin = h + x_edge ; agg[col] += hin[row]
    add_ew_k<<<EW(NH)>>>(h, x_edge, hin, NH);
    fill_f32_k<<<EW(NH)>>>(agg, 0.0f, NH);
    scatter_agg_k<<<EW((size_t)NEDGES * (HID / 4))>>>(hin, row, col, agg, NEDGES);

    // g = (1+eps)*hin + agg  -> bf16
    combine_gin_k<<<EW(NH)>>>(hin, agg, eps, g_bf, NH);

    // MLP1 -> BN -> ReLU (bf16 out)
    transpose_w_bf16_k<<<EW((size_t)HID * HID)>>>(W1, Wt, HID, HID, HID, HID);
    launch_gemm(g_bf, Wt, b1, tmp1, nullptr, nullptr, nullptr, NNODES, HID, HID, 0, stream);
    fill_f32_k<<<EW(2 * HID)>>>(bnsums, 0.0f, 2 * HID);
    bn_stats_k<<<cdiv(NNODES, 64), HID, 0, stream>>>(tmp1, bnsums, NNODES, HID, 64);
    bn_finalize_k<<<1, HID, 0, stream>>>(bnsums, g1, be1, bnss, NNODES, HID);
    bn_apply_k<<<EW(NH)>>>(tmp1, bnss, nullptr, g_bf, HID, NH, 1);

    // MLP2 -> BN -> ReLU (fp32 out = h)
    transpose_w_bf16_k<<<EW((size_t)HID * HID)>>>(W2, Wt, HID, HID, HID, HID);
    launch_gemm(g_bf, Wt, b2, tmp1, nullptr, nullptr, nullptr, NNODES, HID, HID, 0, stream);
    fill_f32_k<<<EW(2 * HID)>>>(bnsums, 0.0f, 2 * HID);
    bn_stats_k<<<cdiv(NNODES, 64), HID, 0, stream>>>(tmp1, bnsums, NNODES, HID, 64);
    bn_finalize_k<<<1, HID, 0, stream>>>(bnsums, bng, bnb, bnss, NNODES, HID);
    bn_apply_k<<<EW(NH)>>>(tmp1, bnss, h, nullptr, HID, NH, 1);
  }

  // ---- pooling ------------------------------------------------------------
  fill_f32_k<<<EW(NGRAPH)>>>(counts, 0.0f, NGRAPH);
  fill_f32_k<<<EW((size_t)NGRAPH * HID)>>>(psum, 0.0f, (size_t)NGRAPH * HID);
  fill_f32_k<<<EW((size_t)NGRAPH * HID)>>>((float*)pmax, 0.0f, (size_t)NGRAPH * HID);
  count_nodes_k<<<EW(NNODES)>>>(batch, counts, NNODES);
  pool_sum_max_k<<<EW((size_t)NNODES * HID)>>>(h, batch, psum, pmax, NNODES);
  pool_to_z_k<<<EW((size_t)NGRAPH * HID)>>>(psum, pmax, counts, z_bf, NGRAPH);

  // ---- head ---------------------------------------------------------------
  // y1 = z @ out_W1 + out_b1 ; BN ; ReLU -> bf16
  transpose_w_bf16_k<<<EW((size_t)HID * (2 * HID))>>>(out_W1, Wt, 2 * HID, HID, 2 * HID, HID);
  launch_gemm(z_bf, Wt, out_b1, y1, nullptr, nullptr, nullptr, NGRAPH, 2 * HID, HID, 0, stream);
  fill_f32_k<<<EW(2 * HID)>>>(bnsums, 0.0f, 2 * HID);
  bn_stats_k<<<cdiv(NGRAPH, 64), HID, 0, stream>>>(y1, bnsums, NGRAPH, HID, 64);
  bn_finalize_k<<<1, HID, 0, stream>>>(bnsums, out_g, out_be, bnss, NGRAPH, HID);
  bn_apply_k<<<EW((size_t)NGRAPH * HID)>>>(y1, bnss, nullptr, zb2, HID, (size_t)NGRAPH * HID, 1);

  // out = zb2 @ out_W2 + out_b2  (N padded 12 -> 128 for the 128-wide tiling)
  pad_bias_k<<<1, 128, 0, stream>>>(out_b2, bias128, NTASK, 128);
  transpose_w_bf16_k<<<EW((size_t)128 * HID)>>>(out_W2, Wt, HID, NTASK, HID, 128);
  launch_gemm(zb2, Wt, bias128, outpad, nullptr, nullptr, nullptr, NGRAPH, HID, 128, 0, stream);
  copy_out_k<<<EW(NGRAPH * NTASK)>>>(outpad, out, NGRAPH, NTASK, 128);
}
